// NNGConv_Model_2783138808453
// MI455X (gfx1250) — matrix operand order repaired and checked by reading, not compiled, not run on hardware
//
#include <hip/hip_runtime.h>
#include <hip/hip_bf16.h>

typedef __bf16 v16bf __attribute__((ext_vector_type(16)));
typedef float  v8f   __attribute__((ext_vector_type(8)));

union ABFrag { unsigned u[8]; v16bf v; };

__device__ __forceinline__ unsigned short f2bf(float f) {
  __bf16 h = (__bf16)f;
  return __builtin_bit_cast(unsigned short, h);
}
__device__ __forceinline__ unsigned pack2bf(float a, float b) {
  return (unsigned)f2bf(a) | ((unsigned)f2bf(b) << 16);
}

// ---------------------------------------------------------------------------
// Kernel 1: fused edge_nn + per-edge matvec as [E,160]x[160,64] bf16 WMMA GEMM,
// scatter-add messages into agg[(N+1),64] and degree into deg[N+1].
// Row N of agg / deg is a dummy sink for tail-tile invalid edges so the
// scatter is branch-free (no exec-mask churn around the atomics).
// Block = 128 threads = 4 waves; each wave owns 16-edge tiles (grid-stride).
// B matrix (20 fragments, 160 VGPRs) is register-resident per wave.
// ---------------------------------------------------------------------------
#define KX 160   // 136 (=17*8) useful K, padded to 5 chunks of 32
__global__ __launch_bounds__(128)
void edge_msg_wmma(const float* __restrict__ x, const int* __restrict__ ei,
                   const float* __restrict__ ea,
                   const float* __restrict__ w_e1, const float* __restrict__ b_e1,
                   const float* __restrict__ w_e2, const float* __restrict__ b_e2,
                   float* __restrict__ agg, float* __restrict__ deg,
                   int N, int E) {
  __shared__ __align__(16) unsigned short sBt[64 * KX];      // B^T [h][k], bf16 bits
  __shared__ __align__(16) unsigned short sU[4][16 * KX];    // per-wave u tile
  __shared__ float sWe1[8 * 16];
  __shared__ float sBe1[16];
  __shared__ __align__(16) int sDst[4][16];

  const int tid = threadIdx.x;

  // Stage B^T: Bt[h][k*8+f] = w_e2[k, f*64+h] (k<16), b_e2[f*64+h] (k==16), 0 pad
  for (int i = tid; i < 64 * KX; i += blockDim.x) {
    int h = i / KX, j = i % KX;
    float v = 0.f;
    if (j < 128)      { int k = j >> 3, f = j & 7; v = w_e2[k * 512 + f * 64 + h]; }
    else if (j < 136) { int f = j - 128;           v = b_e2[f * 64 + h]; }
    sBt[h * KX + j] = f2bf(v);
  }
  for (int i = tid; i < 128; i += blockDim.x) sWe1[i] = w_e1[i];
  if (tid < 16) sBe1[tid] = b_e1[tid];
  __syncthreads();

  const int wave = tid >> 5;
  const int lane = tid & 31;
  const int m    = lane & 15;            // row within 16-edge tile
  const int koff = (lane >> 4) * 8;      // A-layout K offset for upper half-wave

  // ---- hoist all 20 B fragments (5 K-chunks x 4 N-tiles) into registers ----
  ABFrag bfr[5][4];
#pragma unroll
  for (int c = 0; c < 5; ++c) {
    const int kb0 = c * 32 + (lane >> 4) * 8;
#pragma unroll
    for (int nt = 0; nt < 4; ++nt) {
      const unsigned short* Brow = &sBt[(nt * 16 + m) * KX];
#pragma unroll
      for (int v = 0; v < 8; ++v) bfr[c][nt].u[v] = *(const unsigned*)&Brow[kb0 + ((v & 4) ? 16 : 0) + (v & 3) * 2];
    }
  }

  // zero this wave's K-pad region once (wave-local, no barrier needed)
  if (lane < 16) {
    unsigned short* up = &sU[wave][lane * KX];
#pragma unroll
    for (int j = 136; j < KX; j += 2) *(unsigned*)&up[j] = 0u;
  }

  const int ntiles = (E + 15) >> 4;
  const int nwaves = gridDim.x * 4;
  for (int t = blockIdx.x * 4 + wave; t < ntiles; t += nwaves) {
    const int e0 = t << 4;

    // prefetch next tile's edge data (global_prefetch_b8)
    {
      long ne = (long)(t + nwaves) << 4;
      if (ne < E) {
        __builtin_prefetch(ea + ne * 8, 0, 1);
        __builtin_prefetch(ei + ne, 0, 1);
      }
    }

    // ---- edge prep: lanes 0..15, one edge each -> bf16 u row in LDS ----
    if (lane < 16) {
      int  e  = e0 + lane;
      bool ok = e < E;
      int  ec = ok ? e : 0;
      const float4* ear = (const float4*)(ea + (size_t)ec * 8);
      float4 ev0 = ear[0], ev1 = ear[1];
      float eav[8] = {ev0.x, ev0.y, ev0.z, ev0.w, ev1.x, ev1.y, ev1.z, ev1.w};
      int src = ei[ec], dst = ei[E + ec];
      const float4* xr = (const float4*)(x + (size_t)src * 8);
      float4 xv0 = xr[0], xv1 = xr[1];
      float xs[8] = {xv0.x, xv0.y, xv0.z, xv0.w, xv1.x, xv1.y, xv1.z, xv1.w};
      if (!ok) {
#pragma unroll
        for (int f = 0; f < 8; ++f) xs[f] = 0.f;
      }
      unsigned short* up = &sU[wave][lane * KX];
      // h_e[k] computed inline and consumed immediately (no live h1[16] array)
#pragma unroll
      for (int k = 0; k < 16; ++k) {
        float a = sBe1[k];
#pragma unroll
        for (int i = 0; i < 8; ++i) a += eav[i] * sWe1[i * 16 + k];
        float hk = fmaxf(a, 0.f);
#pragma unroll
        for (int f = 0; f < 8; f += 2)
          *(unsigned*)&up[k * 8 + f] = pack2bf(hk * xs[f], hk * xs[f + 1]);
      }
#pragma unroll
      for (int f = 0; f < 8; f += 2)   // h_e[16] = 1 fold of b_e2
        *(unsigned*)&up[128 + f] = pack2bf(xs[f], xs[f + 1]);
      sDst[wave][lane] = ok ? dst : N;          // N = dummy sink row
      atomicAdd(&deg[ok ? dst : N], 1.0f);      // branch-free
    }
    // wave-level LDS ordering: writes above feed cross-lane reads below
    asm volatile("s_wait_dscnt 0" ::: "memory");

    // ---- 5 K-chunks x 4 N-tiles of v_wmma_f32_16x16x32_bf16 (B in regs) ----
    v8f acc[4] = {};
    const unsigned short* Urow = &sU[wave][m * KX];
#pragma unroll
    for (int c = 0; c < 5; ++c) {
      const int Kb = c * 32;
      ABFrag a;
#pragma unroll
      for (int v = 0; v < 8; ++v) {
        int k = Kb + ((v & 4) ? 16 : 0) + koff + (v & 3) * 2;
        a.u[v] = *(const unsigned*)&Urow[k];
      }
#pragma unroll
      for (int nt = 0; nt < 4; ++nt) {
        acc[nt] = __builtin_amdgcn_wmma_f32_16x16x32_bf16(
            false, a.v, false, bfr[c][nt].v, (short)0, acc[nt], false, false);
      }
    }

    // ---- branch-free scatter: lane owns rows rbase..rbase+7, column m ----
    const int rbase = (lane >> 4) * 8;
    int4 d0 = *(const int4*)&sDst[wave][rbase];
    int4 d1 = *(const int4*)&sDst[wave][rbase + 4];
    int drow[8] = {d0.x, d0.y, d0.z, d0.w, d1.x, d1.y, d1.z, d1.w};
#pragma unroll
    for (int nt = 0; nt < 4; ++nt) {
#pragma unroll
      for (int r = 0; r < 8; ++r) {
        atomicAdd(&agg[(size_t)drow[r] * 64 + nt * 16 + m], acc[nt][r]);
      }
    }
  }
}

// ---------------------------------------------------------------------------
// Kernel 2: BN statistics over h_pre = x@root + agg/deg + bias  (recompute trick)
// ---------------------------------------------------------------------------
__global__ void node_stats(const float* __restrict__ x, const float* __restrict__ root,
                           const float* __restrict__ cbias, const float* __restrict__ agg,
                           const float* __restrict__ deg,
                           float* __restrict__ bnsum, float* __restrict__ bnsq, int N) {
  __shared__ float sv[256], sq[256];
  int h = threadIdx.x & 63;
  int n = blockIdx.x * 4 + (threadIdx.x >> 6);
  float val = 0.f;
  if (n < N) {
    float a = cbias[h];
    const float* xr = x + (size_t)n * 8;
#pragma unroll
    for (int f = 0; f < 8; ++f) a += xr[f] * root[f * 64 + h];
    val = a + agg[(size_t)n * 64 + h] / fmaxf(deg[n], 1.f);
  }
  sv[threadIdx.x] = val; sq[threadIdx.x] = val * val;
  __syncthreads();
  if (threadIdx.x < 64) {
    float s = sv[h] + sv[h + 64] + sv[h + 128] + sv[h + 192];
    float q = sq[h] + sq[h + 64] + sq[h + 128] + sq[h + 192];
    atomicAdd(&bnsum[h], s);
    atomicAdd(&bnsq[h], q);
  }
}

// ---------------------------------------------------------------------------
// Kernel 3: normalize + relu + per-graph mean/max pooling (float atomicMax via
// uint ordering; valid since y >= 0 and gmax zero-initialized)
// ---------------------------------------------------------------------------
__global__ void node_norm_pool(const float* __restrict__ x, const float* __restrict__ root,
                               const float* __restrict__ cbias, const float* __restrict__ agg,
                               const float* __restrict__ deg, const float* __restrict__ bnsum,
                               const float* __restrict__ bnsq, const float* __restrict__ gbn,
                               const float* __restrict__ bbn, const int* __restrict__ batch,
                               float* __restrict__ gs, float* __restrict__ gmax,
                               float* __restrict__ cnt, int N) {
  int h = threadIdx.x & 63;
  int n = blockIdx.x * 4 + (threadIdx.x >> 6);
  if (n >= N) return;
  float a = cbias[h];
  const float* xr = x + (size_t)n * 8;
#pragma unroll
  for (int f = 0; f < 8; ++f) a += xr[f] * root[f * 64 + h];
  float val = a + agg[(size_t)n * 64 + h] / fmaxf(deg[n], 1.f);
  float inv = 1.f / (float)N;
  float mean = bnsum[h] * inv;
  float var  = bnsq[h] * inv - mean * mean;
  float y = fmaxf((val - mean) * rsqrtf(var + 1e-5f) * gbn[h] + bbn[h], 0.f);
  int g = batch[n];
  atomicAdd(&gs[g * 64 + h], y);
  atomicMax((unsigned int*)&gmax[g * 64 + h], __float_as_uint(y));
  if (h == 0) atomicAdd(&cnt[g], 1.f);
}

// Kernel 4: g = concat(mean, max)  [64 graphs][128]
__global__ void build_g(const float* __restrict__ gs, const float* __restrict__ gmax,
                        const float* __restrict__ cnt, float* __restrict__ g) {
  int i = blockIdx.x * blockDim.x + threadIdx.x;
  if (i >= 64 * 64) return;
  int gi = i >> 6, h = i & 63;
  float c = fmaxf(cnt[gi], 1.f);
  g[gi * 128 + h]      = gs[i] / c;
  g[gi * 128 + 64 + h] = gmax[i];
}

// Kernel 5: MLP layer with training-mode BN + relu. One thread per output
// feature column -> BN stats over the 64 graph rows are thread-local.
__global__ void mlp_bn_relu(const float* __restrict__ in, const float* __restrict__ W,
                            const float* __restrict__ bias, const float* __restrict__ gamma,
                            const float* __restrict__ beta, float* __restrict__ out,
                            int K, int Nf) {
  int col = blockIdx.x * blockDim.x + threadIdx.x;
  if (col >= Nf) return;
  float v[64], s = 0.f, sq = 0.f;
  for (int r = 0; r < 64; ++r) {
    float a = bias[col];
    const float* ir = in + r * K;
    for (int k = 0; k < K; ++k) a += ir[k] * W[k * Nf + col];
    v[r] = a; s += a; sq += a * a;
  }
  float mean = s * (1.f / 64.f);
  float var  = sq * (1.f / 64.f) - mean * mean;
  float rs = rsqrtf(var + 1e-5f);
  float ga = gamma[col], be = beta[col];
  for (int r = 0; r < 64; ++r)
    out[r * Nf + col] = fmaxf((v[r] - mean) * rs * ga + be, 0.f);
}

// Kernel 6: final linear [64,64] x [64,10]
__global__ void final_out(const float* __restrict__ in, const float* __restrict__ W,
                          const float* __restrict__ b, float* __restrict__ out) {
  int i = blockIdx.x * blockDim.x + threadIdx.x;
  if (i >= 64 * 10) return;
  int r = i / 10, c = i % 10;
  float a = b[c];
#pragma unroll
  for (int k = 0; k < 64; ++k) a += in[r * 64 + k] * W[k * 10 + c];
  out[i] = a;
}

// ---------------------------------------------------------------------------
extern "C" void kernel_launch(void* const* d_in, const int* in_sizes, int n_in,
                              void* d_out, int out_size, void* d_ws, size_t ws_size,
                              hipStream_t stream) {
  const float* x      = (const float*)d_in[0];
  const int*   ei     = (const int*)  d_in[1];
  const float* ea     = (const float*)d_in[2];
  const int*   batch  = (const int*)  d_in[3];
  const float* w_e1   = (const float*)d_in[4];
  const float* b_e1   = (const float*)d_in[5];
  const float* w_e2   = (const float*)d_in[6];
  const float* b_e2   = (const float*)d_in[7];
  const float* root   = (const float*)d_in[8];
  const float* cbias  = (const float*)d_in[9];
  const float* g_bnc  = (const float*)d_in[10];
  const float* b_bnc  = (const float*)d_in[11];
  const float* w1 = (const float*)d_in[12]; const float* b1 = (const float*)d_in[13];
  const float* g1 = (const float*)d_in[14]; const float* be1 = (const float*)d_in[15];
  const float* w2 = (const float*)d_in[16]; const float* b2 = (const float*)d_in[17];
  const float* g2 = (const float*)d_in[18]; const float* be2 = (const float*)d_in[19];
  const float* w3 = (const float*)d_in[20]; const float* b3 = (const float*)d_in[21];
  const float* g3 = (const float*)d_in[22]; const float* be3 = (const float*)d_in[23];
  const float* wout = (const float*)d_in[24]; const float* bout = (const float*)d_in[25];

  const int N = in_sizes[3];
  const int E = in_sizes[1] / 2;

  // workspace layout (floats); agg/deg have a dummy sink row N
  float* ws    = (float*)d_ws;
  float* agg   = ws;                            // (N+1)*64
  float* deg   = agg + (size_t)(N + 1) * 64;    // N+1
  float* bnsum = deg + (N + 1);                 // 64
  float* bnsq  = bnsum + 64;                    // 64
  float* gs    = bnsq + 64;                     // 64*64
  float* gmax  = gs + 64 * 64;                  // 64*64
  float* cnt   = gmax + 64 * 64;                // 64
  float* g     = cnt + 64;                      // 64*128
  float* a1    = g  + 64 * 128;                 // 64*256
  float* a2    = a1 + 64 * 256;                 // 64*128
  float* a3    = a2 + 64 * 128;                 // 64*64

  size_t zero_floats = (size_t)(N + 1) * 64 + (N + 1) + 64 + 64
                     + 64 * 64 + 64 * 64 + 64;
  hipMemsetAsync(d_ws, 0, zero_floats * sizeof(float), stream);

  // 1) fused edge GEMM + scatter (persistent waves)
  edge_msg_wmma<<<1024, 128, 0, stream>>>(x, ei, ea, w_e1, b_e1, w_e2, b_e2,
                                          agg, deg, N, E);
  // 2) BN stats
  int nb = (N + 3) / 4;
  node_stats<<<nb, 256, 0, stream>>>(x, root, cbias, agg, deg, bnsum, bnsq, N);
  // 3) normalize + relu + pooling
  node_norm_pool<<<nb, 256, 0, stream>>>(x, root, cbias, agg, deg, bnsum, bnsq,
                                         g_bnc, b_bnc, batch, gs, gmax, cnt, N);
  // 4) assemble g
  build_g<<<(64 * 64 + 255) / 256, 256, 0, stream>>>(gs, gmax, cnt, g);
  // 5) MLP
  mlp_bn_relu<<<(256 + 63) / 64, 64, 0, stream>>>(g,  w1, b1, g1, be1, a1, 128, 256);
  mlp_bn_relu<<<(128 + 63) / 64, 64, 0, stream>>>(a1, w2, b2, g2, be2, a2, 256, 128);
  mlp_bn_relu<<<(64  + 63) / 64, 64, 0, stream>>>(a2, w3, b3, g3, be3, a3, 128, 64);
  // 6) output head
  final_out<<<(640 + 255) / 256, 256, 0, stream>>>(a3, wout, bout, (float*)d_out);
}
